// SwinMambaBlock_68315749810366
// MI455X (gfx1250) — matrix operand order repaired and checked by reading, not compiled, not run on hardware
//
#include <hip/hip_runtime.h>
#include <cstdint>
#include <cstddef>

// ---------------------------------------------------------------------------
// SwinMambaBlock for MI455X (gfx1250, wave32, WMMA).
// All dense projections run through a bf16 WMMA GEMM (v_wmma_f32_16x16x32_bf16,
// fp32 accum). Tile: 128x128 per block, 32x64 per wave -> 8 WMMAs per wave per
// K-step with B-fragment reuse (~22 FLOP per LDS byte; weights live in L2, so
// LDS read bandwidth is the binding constraint, not HBM's 23.3 TB/s).
// Elementwise / norm passes are HBM-bound (~1us per 26MB tensor). Window
// attention (49x49) and the sequential Mamba scan are latency/VALU work.
// ---------------------------------------------------------------------------

#define T_TOK 50176      // B*H*W tokens
#define HW    12544      // H*W
#define CH    128
#define IMG   112
#define PERB  1605632L   // CH*HW elements per batch

typedef __attribute__((ext_vector_type(16))) __bf16 v16bf;
typedef __attribute__((ext_vector_type(8)))  float  v8f;
union FragBF { unsigned int u[8]; v16bf v; };

__device__ __forceinline__ unsigned short f2bf(float f) {
  union { float f; unsigned int u; } x; x.f = f;
  unsigned int r = x.u + 0x7FFFu + ((x.u >> 16) & 1u);   // round-to-nearest-even
  return (unsigned short)(r >> 16);
}
__device__ __forceinline__ float gelu_f(float x) {
  return 0.5f * x * (1.f + erff(x * 0.70710678118654752f));
}
__device__ __forceinline__ float clip10(float x) { return fminf(fmaxf(x, -10.f), 10.f); }
__device__ __forceinline__ float silu_f(float x) { return x / (1.f + __expf(-x)); }

// ---------------------------------------------------------------------------
// Per-batch GroupNorm(1,C) statistics: two-pass, fp64 partials for accuracy.
// ---------------------------------------------------------------------------
__global__ __launch_bounds__(256) void stats_partial_kernel(
    const float* __restrict__ x, double* __restrict__ partial) {
  int b = blockIdx.x >> 6;        // 64 blocks per batch
  int blk = blockIdx.x & 63;
  const float* p = x + (size_t)b * PERB;
  double s = 0.0, ss = 0.0;
  for (long i = (long)blk * 256 + threadIdx.x; i < PERB; i += 64L * 256L) {
    float v = p[i]; s += v; ss += (double)v * v;
  }
  __shared__ double sh[256], sq[256];
  sh[threadIdx.x] = s; sq[threadIdx.x] = ss; __syncthreads();
  for (int o = 128; o > 0; o >>= 1) {
    if ((int)threadIdx.x < o) { sh[threadIdx.x] += sh[threadIdx.x + o]; sq[threadIdx.x] += sq[threadIdx.x + o]; }
    __syncthreads();
  }
  if (threadIdx.x == 0) {
    partial[(size_t)(b * 64 + blk) * 2]     = sh[0];
    partial[(size_t)(b * 64 + blk) * 2 + 1] = sq[0];
  }
}
__global__ __launch_bounds__(64) void stats_final_kernel(
    const double* __restrict__ partial, float* __restrict__ stats) {
  int b = blockIdx.x, t = threadIdx.x;
  __shared__ double sh[64], sq[64];
  sh[t] = partial[(size_t)(b * 64 + t) * 2];
  sq[t] = partial[(size_t)(b * 64 + t) * 2 + 1];
  __syncthreads();
  for (int o = 32; o > 0; o >>= 1) {
    if (t < o) { sh[t] += sh[t + o]; sq[t] += sq[t + o]; }
    __syncthreads();
  }
  if (t == 0) {
    double mu = sh[0] / (double)PERB;
    double var = sq[0] / (double)PERB - mu * mu;
    stats[b * 2]     = (float)mu;
    stats[b * 2 + 1] = (float)(1.0 / sqrt(var + 1e-5));
  }
}

// ---------------------------------------------------------------------------
// GroupNorm elementwise (same layout in/out). epi: 0=none 1=gelu 2=clip
// layout: 0=NCHW 1=NHWC (only affects channel extraction)
// ---------------------------------------------------------------------------
__global__ __launch_bounds__(256) void gn_elem_kernel(
    const float* __restrict__ src, float* __restrict__ dst,
    const float* __restrict__ gma, const float* __restrict__ bta,
    const float* __restrict__ stats, int nhwc, int epi) {
  size_t idx = (size_t)blockIdx.x * 256 + threadIdx.x;
  int b = (int)(idx / PERB);
  int c = nhwc ? (int)(idx & 127) : (int)((idx / HW) & 127);
  float v = src[idx];
  v = (v - stats[b * 2]) * stats[b * 2 + 1] * gma[c] + bta[c];
  if (epi == 1) v = gelu_f(v);
  else if (epi == 2) v = clip10(v);
  dst[idx] = v;
}

// ---------------------------------------------------------------------------
// GroupNorm + NCHW->NHWC transpose (LDS-tiled, coalesced both sides).
// ---------------------------------------------------------------------------
__global__ __launch_bounds__(256) void gn_transpose_kernel(
    const float* __restrict__ src, float* __restrict__ dst,
    const float* __restrict__ gma, const float* __restrict__ bta,
    const float* __restrict__ stats, int doClip) {
  __shared__ float tile[32][33];
  int bid = blockIdx.x;
  int ttile = bid % 392;
  int ctile = (bid / 392) & 3;
  int b = bid / (392 * 4);
  int tx = threadIdx.x & 31, ty = threadIdx.x >> 5;
  float mu = stats[b * 2], rs = stats[b * 2 + 1];
#pragma unroll
  for (int i = 0; i < 4; ++i) {
    int c = ctile * 32 + ty + i * 8;
    float v = src[((size_t)b * CH + c) * HW + ttile * 32 + tx];
    v = (v - mu) * rs * gma[c] + bta[c];
    if (doClip) v = clip10(v);
    tile[ty + i * 8][tx] = v;
  }
  __syncthreads();
#pragma unroll
  for (int i = 0; i < 4; ++i) {
    int t = ttile * 32 + ty + i * 8;
    dst[((size_t)b * HW + t) * CH + ctile * 32 + tx] = tile[tx][ty + i * 8];
  }
}

// ---------------------------------------------------------------------------
// NHWC -> NCHW transpose with optional NCHW residual add + clip.
// ---------------------------------------------------------------------------
__global__ __launch_bounds__(256) void nhwc2nchw_kernel(
    const float* __restrict__ src, const float* __restrict__ resNCHW,
    float* __restrict__ dst, int doClip) {
  __shared__ float tile[32][33];
  int bid = blockIdx.x;
  int ttile = bid % 392;
  int ctile = (bid / 392) & 3;
  int b = bid / (392 * 4);
  int tx = threadIdx.x & 31, ty = threadIdx.x >> 5;
#pragma unroll
  for (int i = 0; i < 4; ++i) {
    int t = ttile * 32 + ty + i * 8;
    tile[ty + i * 8][tx] = src[((size_t)b * HW + t) * CH + ctile * 32 + tx];
  }
  __syncthreads();
#pragma unroll
  for (int i = 0; i < 4; ++i) {
    int c = ctile * 32 + ty + i * 8;
    int t = ttile * 32 + tx;
    size_t oi = ((size_t)b * CH + c) * HW + t;
    float v = tile[tx][ty + i * 8];
    if (resNCHW) v += resNCHW[oi];
    if (doClip) v = clip10(v);
    dst[oi] = v;
  }
}

// ---------------------------------------------------------------------------
// Wave-per-token LayerNorm over C=128, fused with cyclic shift (roll -s).
// out[(i,j)] = LN(src[(i+shift)%112, (j+shift)%112]) * g + b
// ---------------------------------------------------------------------------
__global__ __launch_bounds__(256) void ln_shift_kernel(
    const float* __restrict__ src, float* __restrict__ dst,
    const float* __restrict__ g, const float* __restrict__ bt, int shift) {
  int token = blockIdx.x * 8 + (threadIdx.x >> 5);
  int lane = threadIdx.x & 31;
  int b = token / HW, hw = token - b * HW;
  int i = hw / IMG, j = hw - i * IMG;
  int si = (i + shift) % IMG, sj = (j + shift) % IMG;
  size_t sidx = ((size_t)b * HW + si * IMG + sj) * CH + lane * 4;
  float4 v = *(const float4*)(src + sidx);
  float s = v.x + v.y + v.z + v.w;
  for (int o = 16; o >= 1; o >>= 1) s += __shfl_xor(s, o, 32);
  float mu = s * (1.f / 128.f);
  float dx = v.x - mu, dy = v.y - mu, dz = v.z - mu, dw = v.w - mu;
  float q = dx * dx + dy * dy + dz * dz + dw * dw;
  for (int o = 16; o >= 1; o >>= 1) q += __shfl_xor(q, o, 32);
  float rstd = rsqrtf(q * (1.f / 128.f) + 1e-5f);
  int c = lane * 4;
  float4 gg = *(const float4*)(g + c);
  float4 bb = *(const float4*)(bt + c);
  float4 o4;
  o4.x = dx * rstd * gg.x + bb.x;
  o4.y = dy * rstd * gg.y + bb.y;
  o4.z = dz * rstd * gg.z + bb.z;
  o4.w = dw * rstd * gg.w + bb.w;
  *(float4*)(dst + (size_t)token * CH + c) = o4;
}

// ---------------------------------------------------------------------------
// out[(i,j)] = g[(i,j)] + proj[(i-3)%112, (j-3)%112]   (reverse roll + residual)
// ---------------------------------------------------------------------------
__global__ __launch_bounds__(256) void shift_add_kernel(
    const float* __restrict__ g, const float* __restrict__ proj,
    float* __restrict__ out) {
  size_t idx = (size_t)blockIdx.x * 256 + threadIdx.x;
  int c = (int)(idx & 127);
  size_t tok = idx >> 7;
  int b = (int)(tok / HW);
  int hw = (int)(tok % HW);
  int i = hw / IMG, j = hw - i * IMG;
  int si = (i + IMG - 3) % IMG, sj = (j + IMG - 3) % IMG;
  out[idx] = g[idx] + proj[((size_t)b * HW + si * IMG + sj) * CH + c];
}

// out = a + b (+ c), optional clip
__global__ __launch_bounds__(256) void add_kernel(
    const float* __restrict__ a, const float* __restrict__ b2,
    const float* __restrict__ c3, float* __restrict__ o, int doClip) {
  size_t idx = (size_t)blockIdx.x * 256 + threadIdx.x;
  float v = a[idx] + b2[idx];
  if (c3) v += c3[idx];
  if (doClip) v = clip10(v);
  o[idx] = v;
}

// ---------------------------------------------------------------------------
// bf16 WMMA GEMM: C[M,N] = A[M,K] * W[K,N] (+bias, epilogue).
// BM=128 BN=128 BK=32, 256 threads = 8 waves laid out 4(m) x 2(n).
// Each wave: 32 rows x 64 cols = 2 A-frags x 4 B-frags -> 8 WMMAs/K-step,
// B-fragments reused across both m-tiles. Fragment gathers follow CDNA5
// ISA 7.12.2 16-bit layouts. float4 staging loads; prefetch next A tile.
// epi: 0 = bias, 1 = bias+GELU, 2 = bias+BatchNorm+GELU
// ---------------------------------------------------------------------------
__global__ __launch_bounds__(256) void gemm_bf16_kernel(
    const float* __restrict__ A, const float* __restrict__ W,
    const float* __restrict__ bias, float* __restrict__ C,
    int M, int N, int K, int epi,
    const float* __restrict__ bng, const float* __restrict__ bnb,
    const float* __restrict__ bnm, const float* __restrict__ bnv) {
  __shared__ unsigned int As32[128 * 17];   // [m][k] bf16 pairs, row stride 17 dwords
  __shared__ unsigned int Bs32[128 * 17];   // [n][k] bf16 pairs (W staged transposed)
  unsigned short* As16 = (unsigned short*)As32;
  unsigned short* Bs16 = (unsigned short*)Bs32;
  const int tid = threadIdx.x;
  const int wave = tid >> 5;
  const int lane = tid & 31;
  const int wrow = wave >> 1;               // 0..3 -> 32-row strip
  const int wcol = wave & 1;                // 0..1 -> 64-col strip
  const int m0 = blockIdx.y * 128;
  const int n0 = blockIdx.x * 128;
  const int row = lane & 15;
  const int hi = lane >> 4;

  v8f zero = {0.f, 0.f, 0.f, 0.f, 0.f, 0.f, 0.f, 0.f};
  v8f acc[2][4] = {{zero, zero, zero, zero}, {zero, zero, zero, zero}};

  const int am = tid >> 3;                  // 0..31
  const int ak = (tid & 7) << 2;            // 0,4,..28
  const int bk = tid >> 5;                  // 0..7
  const int bn = (tid & 31) << 2;           // 0,4,..124  (N is always a multiple of 4)

  for (int kb = 0; kb < K; kb += 32) {
#pragma unroll
    for (int i = 0; i < 4; ++i) {           // A tile 128x32: float4, coalesced on k
      int m = am + i * 32;
      const float4 v = *(const float4*)(A + (size_t)(m0 + m) * K + kb + ak);
      unsigned short* p = As16 + m * 34 + ak;
      p[0] = f2bf(v.x); p[1] = f2bf(v.y); p[2] = f2bf(v.z); p[3] = f2bf(v.w);
    }
    if (kb + 32 < K)                         // prefetch next A tile (global_prefetch_b8)
      __builtin_prefetch(A + (size_t)(m0 + am) * K + kb + 32 + ak, 0, 3);
#pragma unroll
    for (int i = 0; i < 4; ++i) {           // W tile 32x128: float4, coalesced on n
      int k = bk + i * 8;
      float4 v = {0.f, 0.f, 0.f, 0.f};
      if (n0 + bn < N) v = *(const float4*)(W + (size_t)(kb + k) * N + n0 + bn);
      Bs16[(bn + 0) * 34 + k] = f2bf(v.x);
      Bs16[(bn + 1) * 34 + k] = f2bf(v.y);
      Bs16[(bn + 2) * 34 + k] = f2bf(v.z);
      Bs16[(bn + 3) * 34 + k] = f2bf(v.w);
    }
    __syncthreads();
    FragBF af[2];                           // A 16x32: kk=(j<4?0:16)+8*hi+2*(j&3)
#pragma unroll
    for (int mt = 0; mt < 2; ++mt)
#pragma unroll
      for (int j = 0; j < 8; ++j) {
        int kk = ((j < 4) ? 0 : 16) + (hi << 3) + ((j & 3) << 1);
        af[mt].u[j] = As32[(wrow * 32 + mt * 16 + row) * 17 + (kk >> 1)];
      }
#pragma unroll
    for (int nt = 0; nt < 4; ++nt) {        // B 32x16: kk=(j<4?0:16)+8*hi+2*(j&3)
      FragBF bf;
#pragma unroll
      for (int j = 0; j < 8; ++j)
        bf.u[j] = Bs32[(wcol * 64 + nt * 16 + row) * 17 + ((j < 4) ? 0 : 8) + (hi << 2) + (j & 3)];
      acc[0][nt] = __builtin_amdgcn_wmma_f32_16x16x32_bf16(
          false, af[0].v, false, bf.v, (short)0, acc[0][nt], false, false);
      acc[1][nt] = __builtin_amdgcn_wmma_f32_16x16x32_bf16(
          false, af[1].v, false, bf.v, (short)0, acc[1][nt], false, false);
    }
    __syncthreads();
  }

  const int mhi = hi << 3;                  // C: m = r + 8*hi, n = lane&15
#pragma unroll
  for (int nt = 0; nt < 4; ++nt) {
    int n = n0 + wcol * 64 + nt * 16 + row;
    if (n >= N) continue;
    float bsv = bias ? bias[n] : 0.f;
    float g_ = 0.f, b_ = 0.f, m_ = 0.f, r_ = 0.f;
    if (epi == 2) { g_ = bng[n]; b_ = bnb[n]; m_ = bnm[n]; r_ = rsqrtf(bnv[n] + 1e-5f); }
#pragma unroll
    for (int mt = 0; mt < 2; ++mt) {
#pragma unroll
      for (int r = 0; r < 8; ++r) {
        int m = m0 + wrow * 32 + mt * 16 + mhi + r;
        float v = acc[mt][nt][r] + bsv;
        if (epi == 1) v = gelu_f(v);
        else if (epi == 2) v = gelu_f((v - m_) * r_ * g_ + b_);
        C[(size_t)m * N + n] = v;
      }
    }
  }
}

// ---------------------------------------------------------------------------
// Windowed attention: one block per (batch, window, head). 49 tokens, HD=32.
// qkv laid out NHWC (rolled coords): channel = which*128 + head*32 + d.
// Relative-position bias + shift mask computed on the fly.
// ---------------------------------------------------------------------------
__global__ __launch_bounds__(64) void win_attn_kernel(
    const float* __restrict__ qkv, const float* __restrict__ rpb,
    float* __restrict__ out) {
  __shared__ float qs[49 * 32], ks[49 * 32], vs[49 * 32];
  __shared__ float sc[49 * 50];
  __shared__ int rid[49];
  const float SCALE = 0.1767766952966369f;    // 32^-0.5
  int bid = blockIdx.x;
  int head = bid & 3;
  int wrem = bid >> 2;
  int ww = wrem & 15, wh = (wrem >> 4) & 15, b = wrem >> 8;
  int tid = threadIdx.x;
  for (int idx = tid; idx < 49 * 32; idx += 64) {
    int i = idx >> 5, d = idx & 31;
    int gh = wh * 7 + i / 7, gw = ww * 7 + i % 7;
    size_t base = ((size_t)(b * IMG + gh) * IMG + gw) * 384 + head * 32 + d;
    qs[idx] = qkv[base] * SCALE;
    ks[idx] = qkv[base + 128];
    vs[idx] = qkv[base + 256];
  }
  if (tid < 49) {
    int gh = wh * 7 + tid / 7, gw = ww * 7 + tid % 7;
    int rh = (gh < 105) ? 0 : ((gh < 109) ? 1 : 2);
    int rw = (gw < 105) ? 0 : ((gw < 109) ? 1 : 2);
    rid[tid] = rh * 3 + rw;
  }
  __syncthreads();
  int i = tid;
  if (i < 49) {
    int ih = i / 7, iw = i % 7;
    float mx = -1e30f;
    for (int j = 0; j < 49; ++j) {
      float s = 0.f;
      for (int d = 0; d < 32; ++d) s += qs[i * 32 + d] * ks[j * 32 + d];
      int jh = j / 7, jw = j % 7;
      int ridx = (ih - jh + 6) * 13 + (iw - jw + 6);
      s += rpb[ridx * 4 + head];
      if (rid[i] != rid[j]) s -= 100.f;
      sc[i * 50 + j] = s;
      mx = fmaxf(mx, s);
    }
    float sum = 0.f;
    for (int j = 0; j < 49; ++j) {
      float p = __expf(sc[i * 50 + j] - mx);
      sc[i * 50 + j] = p; sum += p;
    }
    float inv = 1.f / sum;
    int gh = wh * 7 + ih, gw = ww * 7 + iw;
    size_t ob = ((size_t)(b * IMG + gh) * IMG + gw) * CH + head * 32;
    for (int d = 0; d < 32; ++d) {
      float a = 0.f;
      for (int j = 0; j < 49; ++j) a += sc[i * 50 + j] * vs[j * 32 + d];
      out[ob + d] = a * inv;
    }
  }
}

// ---------------------------------------------------------------------------
// Causal depthwise conv1d (k=4, pad 3) + SiLU. xi = xz[:, :256].
// ---------------------------------------------------------------------------
__global__ __launch_bounds__(256) void conv_silu_kernel(
    const float* __restrict__ xz, const float* __restrict__ cw,
    const float* __restrict__ cb, float* __restrict__ u) {
  size_t idx = (size_t)blockIdx.x * 256 + threadIdx.x;
  if (idx >= (size_t)T_TOK * 256) return;
  int d = (int)(idx & 255);
  size_t tok = idx >> 8;
  int b = (int)(tok / HW);
  int l = (int)(tok % HW);
  float acc = cb[d];
#pragma unroll
  for (int k = 0; k < 4; ++k) {
    int ls = l - 3 + k;
    if (ls >= 0) acc += cw[d * 4 + k] * xz[((size_t)b * HW + ls) * 512 + d];
  }
  u[idx] = silu_f(acc);
}

// dt = softplus(dt_r @ dt_w + dt_b), dt_r = dbl[:, :8]
__global__ __launch_bounds__(256) void dt_kernel(
    const float* __restrict__ dbl, const float* __restrict__ dtw,
    const float* __restrict__ dtb, float* __restrict__ dt) {
  size_t idx = (size_t)blockIdx.x * 256 + threadIdx.x;
  if (idx >= (size_t)T_TOK * 256) return;
  int d = (int)(idx & 255);
  size_t tok = idx >> 8;
  float acc = dtb[d];
#pragma unroll
  for (int r = 0; r < 8; ++r) acc += dbl[tok * 40 + r] * dtw[r * 256 + d];
  dt[idx] = (acc > 20.f) ? acc : log1pf(__expf(acc));
}

// ---------------------------------------------------------------------------
// Selective-scan: sequential in L per batch, parallel over 256 channels.
// Writes yfull = (y_scan + u*D) * silu(z). B/C loads are wave-uniform -> SMEM.
// ---------------------------------------------------------------------------
__global__ __launch_bounds__(256) void scan_kernel(
    const float* __restrict__ dt, const float* __restrict__ u,
    const float* __restrict__ dbl, const float* __restrict__ xz,
    const float* __restrict__ A_log, const float* __restrict__ Dp,
    float* __restrict__ yout) {
  int b = blockIdx.x;
  int d = threadIdx.x;
  float Ar[16], h[16];
#pragma unroll
  for (int n = 0; n < 16; ++n) { Ar[n] = -__expf(A_log[d * 16 + n]); h[n] = 0.f; }
  float Dv = Dp[d];
  for (int l = 0; l < HW; ++l) {
    size_t tok = (size_t)b * HW + l;
    float dtv = dt[tok * 256 + d];
    float uv = u[tok * 256 + d];
    float zv = xz[tok * 512 + 256 + d];
    const float* bl = dbl + tok * 40;
    float du = dtv * uv;
    float y = 0.f;
#pragma unroll
    for (int n = 0; n < 16; ++n) {
      h[n] = h[n] * __expf(dtv * Ar[n]) + du * bl[8 + n];
      y += h[n] * bl[24 + n];
    }
    y += uv * Dv;
    yout[tok * 256 + d] = y * silu_f(zv);
  }
}

// ---------------------------------------------------------------------------
// Host orchestration
// ---------------------------------------------------------------------------
extern "C" void kernel_launch(void* const* d_in, const int* in_sizes, int n_in,
                              void* d_out, int out_size, void* d_ws, size_t ws_size,
                              hipStream_t stream) {
  if (n_in < 69 || !d_out || !d_ws) return;
  const size_t TC = (size_t)T_TOK * CH;               // 6,422,528 floats
  const size_t need = 8192 + 15 * TC * sizeof(float);
  if (ws_size < need) return;

  // Input order = setup_inputs() recursive insertion order.
  const float* X     = (const float*)d_in[0];
  const float* na1_g = (const float*)d_in[1];
  const float* na1_b = (const float*)d_in[2];
  const float* na2_g = (const float*)d_in[3];
  const float* na2_b = (const float*)d_in[4];
  auto sw = [&](int s, int k) { return (const float*)d_in[5 + s * 15 + k]; };
  enum { SW_GN_G = 0, SW_GN_B, SW_LN1_G, SW_LN1_B, SW_QKV_W, SW_QKV_B, SW_RPB,
         SW_PROJ_W, SW_PROJ_B, SW_LN2_G, SW_LN2_B, SW_FC1_W, SW_FC1_B, SW_FC2_W, SW_FC2_B };
  auto mb = [&](int m, int k) { return (const float*)d_in[35 + m * 17 + k]; };
  enum { MB_GN_G = 0, MB_GN_B, MB_IN_W, MB_CONV_W, MB_CONV_B, MB_XPROJ_W, MB_DT_W,
         MB_DT_B, MB_A_LOG, MB_D, MB_OUT_W, MB_PW_W, MB_PW_B, MB_BN_G, MB_BN_B,
         MB_BN_M, MB_BN_V };

  double* partial = (double*)d_ws;                    // 4*64*2 doubles
  float* stats = (float*)((char*)d_ws + 4096);        // 8 floats
  float* F = (float*)((char*)d_ws + 8192);
  float* bA = F + 0 * TC;        // x_cur (NCHW swin) -> res2 (x1 NCHW)
  float* bB = F + 1 * TC;        // res1 -> mamba block output (NHWC)
  float* bC = F + 2 * TC;        // g / xs (NHWC)
  float* bD = F + 3 * TC;        // t / t2 / dbl
  float* bE = F + 4 * TC;        // qkv / m1 / xz  (4*TC span)
  float* bF = F + 8 * TC;        // attn_out / m2 / ym
  float* bG = F + 9 * TC;        // proj_out ; mamba u (2*TC span) -> pw out
  float* bH = F + 10 * TC;       // h1
  float* bDT = F + 11 * TC;      // dt (2*TC)
  float* bY = F + 13 * TC;       // yfull (2*TC)

  const int EB = (int)(TC / 256);        // elementwise grid (25088)
  const int TR = 4 * 4 * 392;            // transpose grid (6272)
  const int LN = T_TOK / 8;              // LN grid (6272)
  const int CB = (int)((size_t)T_TOK * 256 / 256);   // conv/dt grid (50176)

  auto run_stats = [&](const float* src) {
    stats_partial_kernel<<<256, 256, 0, stream>>>(src, partial);
    stats_final_kernel<<<4, 64, 0, stream>>>(partial, stats);
  };
  auto gemm = [&](const float* A, const float* W, const float* bias, float* C,
                  int N, int K, int epi,
                  const float* g = nullptr, const float* b = nullptr,
                  const float* m = nullptr, const float* v = nullptr) {
    dim3 grid((N + 127) / 128, T_TOK / 128);
    gemm_bf16_kernel<<<grid, 256, 0, stream>>>(A, W, bias, C, T_TOK, N, K, epi, g, b, m, v);
  };

  // ---- Stage A: x = gelu(gn(x, na1)); res1 = x --------------------------
  run_stats(X);
  gn_elem_kernel<<<EB, 256, 0, stream>>>(X, bA, na1_g, na1_b, stats, 0, 1);
  hipMemcpyAsync(bB, bA, TC * sizeof(float), hipMemcpyDeviceToDevice, stream);

  // ---- Swin blocks -------------------------------------------------------
  for (int s = 0; s < 2; ++s) {
    run_stats(bA);
    gn_transpose_kernel<<<TR, 256, 0, stream>>>(bA, bC, sw(s, SW_GN_G), sw(s, SW_GN_B), stats, 0);
    ln_shift_kernel<<<LN, 256, 0, stream>>>(bC, bD, sw(s, SW_LN1_G), sw(s, SW_LN1_B), 3);
    gemm(bD, sw(s, SW_QKV_W), sw(s, SW_QKV_B), bE, 384, 128, 0);
    win_attn_kernel<<<4096, 64, 0, stream>>>(bE, sw(s, SW_RPB), bF);
    gemm(bF, sw(s, SW_PROJ_W), sw(s, SW_PROJ_B), bG, 128, 128, 0);
    shift_add_kernel<<<EB, 256, 0, stream>>>(bC, bG, bH);               // h1 = g + attn
    ln_shift_kernel<<<LN, 256, 0, stream>>>(bH, bD, sw(s, SW_LN2_G), sw(s, SW_LN2_B), 0);
    gemm(bD, sw(s, SW_FC1_W), sw(s, SW_FC1_B), bE, 512, 128, 1);        // +gelu
    gemm(bE, sw(s, SW_FC2_W), sw(s, SW_FC2_B), bF, 128, 512, 0);
    add_kernel<<<EB, 256, 0, stream>>>(bH, bF, bC, bF, 0);              // h1+mlp+g (=2g+attn+mlp)
    nhwc2nchw_kernel<<<TR, 256, 0, stream>>>(bF, nullptr, bA, 1);       // clip -> NCHW
  }
  add_kernel<<<EB, 256, 0, stream>>>(bA, bB, nullptr, bA, 1);           // x = clip(x+res1)
  run_stats(bA);
  gn_elem_kernel<<<EB, 256, 0, stream>>>(bA, bA, na2_g, na2_b, stats, 0, 1);  // x1 = res2 (NCHW)

  // ---- Mamba blocks ------------------------------------------------------
  for (int m = 0; m < 2; ++m) {
    if (m == 0) {
      run_stats(bA);
      gn_transpose_kernel<<<TR, 256, 0, stream>>>(bA, bC, mb(m, MB_GN_G), mb(m, MB_GN_B), stats, 1);
    } else {
      run_stats(bB);
      gn_elem_kernel<<<EB, 256, 0, stream>>>(bB, bC, mb(m, MB_GN_G), mb(m, MB_GN_B), stats, 1, 2);
    }
    gemm(bC, mb(m, MB_IN_W), nullptr, bE, 512, 128, 0);                 // xz
    conv_silu_kernel<<<CB, 256, 0, stream>>>(bE, mb(m, MB_CONV_W), mb(m, MB_CONV_B), bG);  // u
    gemm(bG, mb(m, MB_XPROJ_W), nullptr, bD, 40, 256, 0);               // dbl
    dt_kernel<<<CB, 256, 0, stream>>>(bD, mb(m, MB_DT_W), mb(m, MB_DT_B), bDT);
    scan_kernel<<<4, 256, 0, stream>>>(bDT, bG, bD, bE, mb(m, MB_A_LOG), mb(m, MB_D), bY);
    gemm(bY, mb(m, MB_OUT_W), nullptr, bF, 128, 256, 0);                // ym
    gemm(bF, mb(m, MB_PW_W), mb(m, MB_PW_B), bG, 128, 128, 2,
         mb(m, MB_BN_G), mb(m, MB_BN_B), mb(m, MB_BN_M), mb(m, MB_BN_V));
    add_kernel<<<EB, 256, 0, stream>>>(bC, bG, nullptr, bB, 1);         // clip(res + y) NHWC
  }

  // ---- Final: out = clip(x_mamba + res2) -> NCHW -------------------------
  nhwc2nchw_kernel<<<TR, 256, 0, stream>>>(bB, bA, (float*)d_out, 1);
}
